// MSDeformAttnPixelDecoder_55542517072599
// MI455X (gfx1250) — matrix operand order repaired and checked by reading, not compiled, not run on hardware
//
#include <hip/hip_runtime.h>
#include <hip/hip_bf16.h>

// ---------------------------------------------------------------------------
// Problem constants (from reference)
// ---------------------------------------------------------------------------
#define B_     8
#define C_     256
#define NH_    8
#define NL_    4
#define NP_    4
#define NLAY_  6
#define FFN_   1024
#define QTOT   5440            // 64*64 + 32*32 + 16*16 + 8*8
#define MQ     (B_ * QTOT)     // 43520  (divisible by 128)
#define DH_    32

typedef __attribute__((ext_vector_type(16))) __bf16 v16bf;
typedef __attribute__((ext_vector_type(8)))  float  v8f;
typedef __attribute__((ext_vector_type(4)))  unsigned int v4u;
typedef __attribute__((ext_vector_type(8)))  int    v8i_;
typedef __attribute__((ext_vector_type(4)))  int    v4i_;

__device__ __forceinline__ __bf16 cvt_bf16(float f) {
    union { float f; unsigned u; } a; a.f = f;
    unsigned r = (a.u + 0x7FFFu + ((a.u >> 16) & 1u)) >> 16;   // RNE
    union { unsigned short s; __bf16 b; } o; o.s = (unsigned short)r;
    return o.b;
}

// ---------------------------------------------------------------------------
// bf16 WMMA GEMM:  C[M,N] = A[M,K] @ Bt[N,K]^T + bias (+ReLU)
//   A  : bf16 row-major [M][K]
//   Bt : bf16 row-major [N][K]  (pre-transposed weights)
//   C  : f32 (Cf) or bf16 (Cb) output, one of them non-null
// Block tile 128x128, 8 waves, wave tile 64x32 = 8 x v_wmma_f32_16x16x32_bf16
// per K-step.  A tile staged with b128 copies; B tile fetched by the Tensor
// Data Mover (tensor_load_to_lds + s_wait_tensorcnt).
// Requires M%128==0, N%128==0, K%32==0 (true for all GEMMs here).
// ---------------------------------------------------------------------------
#define LDT_ 40   // padded bf16 row stride: 80B, multiple of 16B

__global__ __launch_bounds__(256) void gemm_bf16_wmma(
    const __bf16* __restrict__ A, const __bf16* __restrict__ Bt,
    const float* __restrict__ bias,
    float* __restrict__ Cf, __bf16* __restrict__ Cb,
    int M, int N, int K, int relu)
{
    __shared__ __bf16 lsA[128 * LDT_];   // [m][k]
    __shared__ __bf16 lsB[128 * LDT_];   // [n][k]

    const int tid  = threadIdx.x;
    const int lane = tid & 31;
    const int wid  = tid >> 5;
    const int wm   = wid >> 2;          // 0..1
    const int wn   = wid & 3;           // 0..3
    const int blockM = blockIdx.x * 128;
    const int blockN = blockIdx.y * 128;
    const int hh   = lane >> 4;         // lane half
    const int rsub = lane & 15;

    v8f acc[4][2] = {};

    // LDS aperture addresses have the byte offset in the low 32 bits
    const unsigned ldsb_addr = (unsigned)(size_t)(&lsB[0]);

    for (int k0 = 0; k0 < K; k0 += 32) {
        // ---- stage A tile 128x32 bf16 with b128 copies ----
        #pragma unroll
        for (int p = 0; p < 2; ++p) {
            int chunk = tid + p * 256;              // 512 chunks of 8 bf16
            int r  = chunk >> 2;
            int c8 = (chunk & 3) * 8;
            uint4 d = *(const uint4*)(A + (size_t)(blockM + r) * K + (k0 + c8));
            *(uint4*)&lsA[r * LDT_ + c8] = d;
        }
        // ---- B tile 128(n) x 32(k) via Tensor Data Mover ----
        if (tid == 0) {
            unsigned long long ga =
                (unsigned long long)(size_t)(Bt + (size_t)blockN * K + k0);
            v4u g0;
            g0[0] = 1u;                                   // count=1, no gather
            g0[1] = ldsb_addr;                            // lds_addr
            g0[2] = (unsigned)ga;                         // global_addr[31:0]
            g0[3] = (unsigned)((ga >> 32) & 0x01FFFFFFull) | (2u << 30); // [56:32] | type=2
            v8i_ g1;
            // data_size=2B, pad_enable, pad_interval=16dw (code 3), pad_amount=4dw (code 3)
            g1[0] = (int)((1u << 16) | (1u << 20) | (3u << 22) | (3u << 25));
            g1[1] = (int)(((unsigned)K & 0xFFFFu) << 16);                 // tensor_dim0 lo16
            g1[2] = (int)((((unsigned)K >> 16) & 0xFFFFu) |               // tensor_dim0 hi16
                          (((unsigned)N & 0xFFFFu) << 16));               // tensor_dim1 lo16
            g1[3] = (int)((((unsigned)N >> 16) & 0xFFFFu) | (32u << 16)); // dim1 hi | tile_dim0=32
            g1[4] = 128;                                                  // tile_dim1=128
            g1[5] = K;                                                    // dim0_stride[31:0]
            g1[6] = (int)(((unsigned)K & 0xFFFFu) << 16);                 // dim1_stride lo16
            g1[7] = (int)(((unsigned)K >> 16) & 0xFFFFu);                 // dim1_stride hi
            v4i_ zz = {0, 0, 0, 0};
            v8i_ z8 = {0, 0, 0, 0, 0, 0, 0, 0};
            __builtin_amdgcn_tensor_load_to_lds(g0, g1, zz, zz, z8, 0);
            __builtin_amdgcn_s_wait_tensorcnt(0);
        }
        if (k0 + 32 < K)
            __builtin_prefetch(A + (size_t)(blockM + (tid >> 1)) * K + (k0 + 32), 0, 1);
        __syncthreads();

        // ---- fragments per ISA 16-bit layouts ----
        union Frag { v16bf v; unsigned u[8]; };
        Frag fa[4], fb[2];
        #pragma unroll
        for (int i = 0; i < 4; ++i) {
            int row = wm * 64 + i * 16 + rsub;
            #pragma unroll
            for (int v = 0; v < 8; ++v) {
                // A 16x32: k = (v<4?0:16) + half*8 + (v%4)*2
                int kk = ((v & 4) << 2) + hh * 8 + (v & 3) * 2;
                fa[i].u[v] = *(const unsigned int*)&lsA[row * LDT_ + kk];
            }
        }
        #pragma unroll
        for (int j = 0; j < 2; ++j) {
            int col = wn * 32 + j * 16 + rsub;
            #pragma unroll
            for (int v = 0; v < 8; ++v) {
                // B 32x16: k = (v<4?0:16) + half*8 + (v%4)*2
                int kk = ((v & 4) << 2) + hh * 8 + (v & 3) * 2;
                fb[j].u[v] = *(const unsigned int*)&lsB[col * LDT_ + kk];
            }
        }
        #pragma unroll
        for (int i = 0; i < 4; ++i)
            #pragma unroll
            for (int j = 0; j < 2; ++j)
                acc[i][j] = __builtin_amdgcn_wmma_f32_16x16x32_bf16(
                    false, fa[i].v, false, fb[j].v,
                    (short)0, acc[i][j], false, false);
        __syncthreads();
    }

    // ---- epilogue ----
    #pragma unroll
    for (int i = 0; i < 4; ++i) {
        int m0 = blockM + wm * 64 + i * 16 + (hh << 3);
        #pragma unroll
        for (int j = 0; j < 2; ++j) {
            int n = blockN + wn * 32 + j * 16 + rsub;
            float bv = bias ? bias[n] : 0.0f;
            #pragma unroll
            for (int r = 0; r < 8; ++r) {
                float val = acc[i][j][r] + bv;
                if (relu) val = fmaxf(val, 0.0f);
                if (Cb) Cb[(size_t)(m0 + r) * N + n] = cvt_bf16(val);
                else    Cf[(size_t)(m0 + r) * N + n] = val;
            }
        }
    }
}

// ---------------------------------------------------------------------------
// Weight transpose + f32 -> bf16:  Wt[n][k] = W[k][n]
// ---------------------------------------------------------------------------
__global__ void wcvt_kernel(const float* __restrict__ W, __bf16* __restrict__ Wt,
                            int K, int N)
{
    long long idx = (long long)blockIdx.x * blockDim.x + threadIdx.x;
    if (idx >= (long long)K * N) return;
    int k = (int)(idx / N), n = (int)(idx % N);
    Wt[(size_t)n * K + k] = cvt_bf16(W[idx]);
}

// f32 -> bf16 elementwise
__global__ void cvt_kernel(const float* __restrict__ x, __bf16* __restrict__ y,
                           long long ntot)
{
    long long idx = (long long)blockIdx.x * blockDim.x + threadIdx.x;
    if (idx >= ntot) return;
    y[idx] = cvt_bf16(x[idx]);
}

// pos = pos_flat + level_embed[level(q)]
__global__ void prep_pos_kernel(const float* __restrict__ pos_flat,
                                const float* __restrict__ lvl_emb,
                                float* __restrict__ pos)
{
    long long idx = (long long)blockIdx.x * blockDim.x + threadIdx.x;
    if (idx >= (long long)MQ * C_) return;
    int col = (int)(idx & (C_ - 1));
    int row = (int)(idx >> 8);
    int qi  = row % QTOT;
    int l   = (qi < 4096) ? 0 : (qi < 5120) ? 1 : (qi < 5376) ? 2 : 3;
    pos[idx] = pos_flat[idx] + lvl_emb[l * C_ + col];
}

// q16 = bf16(out + pos)
__global__ void addq_kernel(const float* __restrict__ a,
                            const float* __restrict__ b,
                            __bf16* __restrict__ y16)
{
    long long idx = (long long)blockIdx.x * blockDim.x + threadIdx.x;
    if (idx >= (long long)MQ * C_) return;
    y16[idx] = cvt_bf16(a[idx] + b[idx]);
}

// ---------------------------------------------------------------------------
// Deformable-attention sampling: one wave per (b, q, head), lane = channel.
// ---------------------------------------------------------------------------
__global__ __launch_bounds__(256) void msda_sample_kernel(
    const float* __restrict__ value,   // (B*Q, 256)
    const float* __restrict__ off,     // (B*Q, 256)
    const float* __restrict__ logit,   // (B*Q, 128)
    __bf16* __restrict__ outp16)       // (B*Q, 256) bf16
{
    const int wv   = blockIdx.x * 8 + (threadIdx.x >> 5);
    const int lane = threadIdx.x & 31;
    const int h  = wv % NH_;
    const int t  = wv / NH_;
    const int qi = t % QTOT;
    const int b  = t / QTOT;

    int H0, W0, loc;
    if (qi < 4096)      { H0 = 64; W0 = 64; loc = qi; }
    else if (qi < 5120) { H0 = 32; W0 = 32; loc = qi - 4096; }
    else if (qi < 5376) { H0 = 16; W0 = 16; loc = qi - 5120; }
    else                { H0 = 8;  W0 = 8;  loc = qi - 5376; }
    const float refx = ((loc % W0) + 0.5f) / (float)W0;
    const float refy = ((loc / W0) + 0.5f) / (float)H0;

    const float* lrow = logit + (size_t)t * (NH_ * NL_ * NP_) + h * 16;
    float lg[16];
    float mx = -1e30f;
    #pragma unroll
    for (int s = 0; s < 16; ++s) { lg[s] = lrow[s]; mx = fmaxf(mx, lg[s]); }
    float sum = 0.0f;
    #pragma unroll
    for (int s = 0; s < 16; ++s) { lg[s] = __expf(lg[s] - mx); sum += lg[s]; }
    const float inv = 1.0f / sum;

    const float* offrow = off + (size_t)t * C_ + h * (NL_ * NP_ * 2);
    const int lstart[4] = {0, 4096, 5120, 5376};
    const int ldim[4]   = {64, 32, 16, 8};

    float acc = 0.0f;
    #pragma unroll
    for (int l = 0; l < 4; ++l) {
        const int Hl = ldim[l], Wl = ldim[l];
        const float* vbase = value + ((size_t)b * QTOT + lstart[l]) * C_ + h * DH_ + lane;
        #pragma unroll
        for (int p = 0; p < 4; ++p) {
            const float ox = offrow[(l * 4 + p) * 2 + 0];
            const float oy = offrow[(l * 4 + p) * 2 + 1];
            const float x = refx * Wl + ox - 0.5f;
            const float y = refy * Hl + oy - 0.5f;
            const float x0 = floorf(x), y0 = floorf(y);
            const float fx = x - x0,   fy = y - y0;
            const int ix0 = (int)x0, iy0 = (int)y0;
            const float aw = lg[l * 4 + p] * inv;
            #pragma unroll
            for (int cc = 0; cc < 4; ++cc) {
                const int dy = cc >> 1, dx = cc & 1;
                const int cy = iy0 + dy, cx = ix0 + dx;
                if (cy >= 0 && cy < Hl && cx >= 0 && cx < Wl) {
                    const float wgt = (dy ? fy : 1.0f - fy) * (dx ? fx : 1.0f - fx);
                    acc += vbase[(size_t)(cy * Wl + cx) * C_] * (wgt * aw);
                }
            }
        }
    }
    outp16[(size_t)t * C_ + h * DH_ + lane] = cvt_bf16(acc);
}

// ---------------------------------------------------------------------------
// y = LayerNorm(x + delta)*g + b ; also emits bf16 copy for downstream GEMMs
// ---------------------------------------------------------------------------
__global__ __launch_bounds__(256) void ln_residual_kernel(
    const float* __restrict__ x, const float* __restrict__ delta,
    const float* __restrict__ g, const float* __restrict__ bb,
    float* __restrict__ y, __bf16* __restrict__ y16)
{
    const int row  = blockIdx.x * 8 + (threadIdx.x >> 5);
    const int lane = threadIdx.x & 31;
    const float* xr = x + (size_t)row * C_;
    const float* dr = delta + (size_t)row * C_;

    float v[8];
    float s = 0.0f;
    #pragma unroll
    for (int i = 0; i < 8; ++i) { v[i] = xr[lane + i * 32] + dr[lane + i * 32]; s += v[i]; }
    #pragma unroll
    for (int o = 16; o >= 1; o >>= 1) s += __shfl_xor(s, o, 32);
    const float mean = s * (1.0f / C_);

    float s2 = 0.0f;
    #pragma unroll
    for (int i = 0; i < 8; ++i) { float d = v[i] - mean; s2 += d * d; }
    #pragma unroll
    for (int o = 16; o >= 1; o >>= 1) s2 += __shfl_xor(s2, o, 32);
    const float rstd = rsqrtf(s2 * (1.0f / C_) + 1e-5f);

    float*  yr  = y   + (size_t)row * C_;
    __bf16* yr16 = y16 + (size_t)row * C_;
    #pragma unroll
    for (int i = 0; i < 8; ++i) {
        int c = lane + i * 32;
        float o = (v[i] - mean) * rstd * g[c] + bb[c];
        yr[c] = o;
        yr16[c] = cvt_bf16(o);
    }
}

// ---------------------------------------------------------------------------
// Host-side orchestration
// ---------------------------------------------------------------------------
extern "C" void kernel_launch(void* const* d_in, const int* in_sizes, int n_in,
                              void* d_out, int out_size, void* d_ws, size_t ws_size,
                              hipStream_t stream)
{
    (void)in_sizes; (void)n_in; (void)out_size; (void)ws_size;
    const float* src_flat  = (const float*)d_in[0];
    const float* pos_flat  = (const float*)d_in[1];
    const float* level_emb = (const float*)d_in[2];
    const float* W_samp    = (const float*)d_in[3];
    const float* b_samp    = (const float*)d_in[4];
    const float* W_attn    = (const float*)d_in[5];
    const float* b_attn    = (const float*)d_in[6];
    const float* W_val     = (const float*)d_in[7];
    const float* b_val     = (const float*)d_in[8];
    const float* W_out     = (const float*)d_in[9];
    const float* b_out     = (const float*)d_in[10];
    const float* g1        = (const float*)d_in[11];
    const float* b1        = (const float*)d_in[12];
    const float* W_fc1     = (const float*)d_in[13];
    const float* b_fc1     = (const float*)d_in[14];
    const float* W_fc2     = (const float*)d_in[15];
    const float* b_fc2     = (const float*)d_in[16];
    const float* g2        = (const float*)d_in[17];
    const float* b2        = (const float*)d_in[18];

    float* out_final = (float*)d_out;

    // workspace layout; ROW = MQ*C f32 elements (44.56 MB)
    const size_t ROW = (size_t)MQ * C_;
    float* ws     = (float*)d_ws;
    float*  pos    = ws;                                  // [0R,1R)
    float*  outb   = ws + ROW;                            // [1R,2R)
    float*  tmp    = ws + 2 * ROW;                        // [2R,3R)
    float*  val    = ws + 3 * ROW;                        // [3R,4R)
    float*  offb   = ws + 4 * ROW;                        // [4R,5R)
    float*  logit  = ws + 5 * ROW;                        // [5R,5.5R)
    __bf16* outb16 = (__bf16*)(ws + 5 * ROW + ROW / 2);   // [5.5R,6R)
    __bf16* qb16   = (__bf16*)(ws + 6 * ROW);             // [6R,6.5R)
    __bf16* attn16 = (__bf16*)(ws + 6 * ROW + ROW / 2);   // [6.5R,7R)
    __bf16* wbuf   = (__bf16*)(ws + 7 * ROW);             // bf16 weights (~9 MB)
    __bf16* hid16  = (__bf16*)val;                        // aliases [3R,5R): MQ x FFN bf16

    const int EW_BLOCKS   = (int)(ROW / 256);             // 43520
    const int LN_BLOCKS   = MQ / 8;                       // 5440
    const int SAMP_BLOCKS = (MQ * NH_) / 8;               // 43520

    // one-time weight transpose+convert (all layers)
    const size_t LW = 65536u * 3 + 32768u + 262144u * 2;  // 753664 per layer
    for (int i = 0; i < NLAY_; ++i) {
        __bf16* wb = wbuf + (size_t)i * LW;
        wcvt_kernel<<<(65536 + 255) / 256, 256, 0, stream>>>(W_val  + (size_t)i * 65536,  wb,          C_,  C_);
        wcvt_kernel<<<(65536 + 255) / 256, 256, 0, stream>>>(W_samp + (size_t)i * 65536,  wb + 65536,  C_,  256);
        wcvt_kernel<<<(32768 + 255) / 256, 256, 0, stream>>>(W_attn + (size_t)i * 32768,  wb + 131072, C_,  128);
        wcvt_kernel<<<(65536 + 255) / 256, 256, 0, stream>>>(W_out  + (size_t)i * 65536,  wb + 163840, C_,  C_);
        wcvt_kernel<<<(262144 + 255) / 256, 256, 0, stream>>>(W_fc1 + (size_t)i * 262144, wb + 229376, C_,  FFN_);
        wcvt_kernel<<<(262144 + 255) / 256, 256, 0, stream>>>(W_fc2 + (size_t)i * 262144, wb + 491520, FFN_, C_);
    }

    // out = src_flat (f32 + bf16 copies); pos = pos_flat + level_embed
    (void)hipMemcpyAsync(outb, src_flat, ROW * sizeof(float),
                         hipMemcpyDeviceToDevice, stream);
    cvt_kernel<<<EW_BLOCKS, 256, 0, stream>>>(src_flat, outb16, (long long)ROW);
    prep_pos_kernel<<<EW_BLOCKS, 256, 0, stream>>>(pos_flat, level_emb, pos);

    for (int i = 0; i < NLAY_; ++i) {
        __bf16* wb  = wbuf + (size_t)i * LW;
        __bf16* WvT = wb;            // [256][256]
        __bf16* WsT = wb + 65536;    // [256][256]
        __bf16* WaT = wb + 131072;   // [128][256]
        __bf16* WoT = wb + 163840;   // [256][256]
        __bf16* W1T = wb + 229376;   // [1024][256]
        __bf16* W2T = wb + 491520;   // [256][1024]

        // q16 = bf16(out + pos)
        addq_kernel<<<EW_BLOCKS, 256, 0, stream>>>(outb, pos, qb16);

        // value = out @ W_val + b_val            (f32 out, for sampling)
        gemm_bf16_wmma<<<dim3(MQ / 128, C_ / 128), 256, 0, stream>>>(
            outb16, WvT, b_val + i * C_, val, nullptr, MQ, C_, C_, 0);
        // off = q @ W_samp + b_samp
        gemm_bf16_wmma<<<dim3(MQ / 128, 256 / 128), 256, 0, stream>>>(
            qb16, WsT, b_samp + i * 256, offb, nullptr, MQ, 256, C_, 0);
        // logits = q @ W_attn + b_attn
        gemm_bf16_wmma<<<dim3(MQ / 128, 128 / 128), 256, 0, stream>>>(
            qb16, WaT, b_attn + i * 128, logit, nullptr, MQ, 128, C_, 0);

        // deformable sampling -> attn16
        msda_sample_kernel<<<SAMP_BLOCKS, 256, 0, stream>>>(val, offb, logit, attn16);

        // proj = attn @ W_out + b_out (f32)
        gemm_bf16_wmma<<<dim3(MQ / 128, C_ / 128), 256, 0, stream>>>(
            attn16, WoT, b_out + i * C_, tmp, nullptr, MQ, C_, C_, 0);
        // out = LN(out + proj)   (f32 + bf16)
        ln_residual_kernel<<<LN_BLOCKS, 256, 0, stream>>>(
            outb, tmp, g1 + i * C_, b1 + i * C_, outb, outb16);

        // hid16 = bf16(relu(out @ W_fc1 + b_fc1))
        gemm_bf16_wmma<<<dim3(MQ / 128, FFN_ / 128), 256, 0, stream>>>(
            outb16, W1T, b_fc1 + i * FFN_, nullptr, hid16, MQ, FFN_, C_, 1);
        // ffn = hid @ W_fc2 + b_fc2 (f32)
        gemm_bf16_wmma<<<dim3(MQ / 128, C_ / 128), 256, 0, stream>>>(
            hid16, W2T, b_fc2 + i * C_, tmp, nullptr, MQ, C_, FFN_, 0);
        // out = LN(out + ffn); final layer writes d_out
        float* dst = (i == NLAY_ - 1) ? out_final : outb;
        ln_residual_kernel<<<LN_BLOCKS, 256, 0, stream>>>(
            outb, tmp, g2 + i * C_, b2 + i * C_, dst, outb16);
    }
}
